// PolicyNet_52321291600490
// MI455X (gfx1250) — matrix operand order repaired and checked by reading, not compiled, not run on hardware
//
#include <hip/hip_runtime.h>
#include <hip/hip_bf16.h>
#include <math.h>

#define L_ 8
#define N_ 32768
#define H_ 404
#define D_ 128
#define F_ 532
#define KCH 17                      // 16 full 32-wide K chunks + masked tail
#define NFULL 16
#define NBUF 6                      // A-tile LDS pipeline depth (5 in flight)
#define WGROWS 128
#define WGS_PER_HOP (N_ / WGROWS)   // 256
#define BWORDS (8 * KCH * 32 * 8)   // 34816 packed-bf16 words of W_fa
#define AROWW 36                    // padded row stride (words) per chunk-row
#define ASLOTW (WGROWS * AROWW)     // 4608 words per A slot (18 KB)
#define AWORDS (NBUF * ASLOTW)      // 27648
#define WMAXW (8 * 128)
#define K1_SMEM_BYTES ((BWORDS + AWORDS + WMAXW) * 4)   // 253952 B

typedef __attribute__((ext_vector_type(16))) __bf16 v16bf;
typedef __attribute__((ext_vector_type(8)))  float  v8f;

union BfVec { unsigned int u[8]; v16bf v; };

__device__ __forceinline__ unsigned int pk_bf16(float x, float y) {
  return (__float_as_uint(y) & 0xFFFF0000u) | (__float_as_uint(x) >> 16);
}

__device__ __forceinline__ void wait_async(int n) {   // n is always a literal
  switch (n) {
    case 0:  asm volatile("s_wait_asynccnt 0x0"  ::: "memory"); break;
    case 4:  asm volatile("s_wait_asynccnt 0x4"  ::: "memory"); break;
    case 8:  asm volatile("s_wait_asynccnt 0x8"  ::: "memory"); break;
    case 12: asm volatile("s_wait_asynccnt 0xc"  ::: "memory"); break;
    case 16: asm volatile("s_wait_asynccnt 0x10" ::: "memory"); break;
    default: asm volatile("s_wait_asynccnt 0x14" ::: "memory"); break;  // 20
  }
}

// ---------------------------------------------------------------------------
// k0: pack W_fa (D,F) fp32 -> bf16 in WMMA B-matrix register order, K padded.
//   lanes 0-15 : K = 2v,2v+1 (v<4), 16+2(v-4),+1 (v>=4) (N = lane) ; lanes 16-31 : same +8
// Flat word index (((c*17 + t)*32 + lane)*8 + v).
// ---------------------------------------------------------------------------
__global__ void k0_pack_wfa(const float* __restrict__ Wfa,
                            unsigned int* __restrict__ Bp) {
  int w = blockIdx.x * blockDim.x + threadIdx.x;
  if (w >= BWORDS) return;
  int v  = w & 7;
  int ln = (w >> 3) & 31;
  int ct = w >> 8;                 // c*KCH + t
  int t  = ct % KCH;
  int c  = ct / KCH;
  int k0 = t * 32 + ((v >= 4) ? 16 : 0) + ((ln >= 16) ? 8 : 0) + 2 * (v & 3);
  int d  = c * 16 + (ln & 15);
  float lo = 0.f, hi = 0.f;
  if (k0 < F_) { lo = Wfa[d * F_ + k0]; hi = Wfa[d * F_ + k0 + 1]; }
  Bp[w] = pk_bf16(lo, hi);
}

// ---------------------------------------------------------------------------
// k1: streaming GEMM hnts = neighbors x W_fa^T (+b_fa), fused per-hop
// column-max, action-row gather, hop-7 row dump.
// One WG (8 waves) = 128 rows of one hop; wave = 16 rows x 128 cols.
//  - B (139 KB) async-staged to LDS once per WG.
//  - A streamed through a 6-slot async LDS pipeline (wave-private rows),
//    5 chunks (80 KB/WGP) in flight to cover HBM latency at 23.3 TB/s.
//  - inner loop: 4x ds A reads + reg-double-buffered ds B reads + 8x wmma.
// ---------------------------------------------------------------------------
__global__ void __launch_bounds__(256)
k1_gemm(const float* __restrict__ neighbors,
        const float* __restrict__ b_fa,
        const int*   __restrict__ action_ids,
        const unsigned int* __restrict__ Bp,
        float* __restrict__ wgmax,      // (L, 256, 128)
        float* __restrict__ hsel,       // (L-1, 128)
        float* __restrict__ hnts7)      // (N, 128)
{
  extern __shared__ unsigned int smem[];        // B | A slots | wavemax
  float* Alds    = (float*)(smem + BWORDS);
  float* wavemax = (float*)(smem + BWORDS + AWORDS);

  const int tid  = threadIdx.x;
  const int lane = tid & 31;
  const int wid  = tid >> 5;
  const int lo   = lane & 15;
  const int hi   = lane >> 4;
  const int hi8  = hi * 8;
  const int l    = blockIdx.x >> 8;
  const int wgi  = blockIdx.x & 255;
  const int n0   = wgi * WGROWS + wid * 16;

  // ---- async-stage packed B into LDS: 34 x b128 per thread ----
  for (int i = tid; i < BWORDS / 4; i += 256) {
    unsigned int loff = (unsigned int)(size_t)(void*)&smem[i * 4];
    const uint4* g = (const uint4*)Bp + i;
    asm volatile("global_load_async_to_lds_b128 %0, %1, off"
                 :: "v"(loff), "v"(g) : "memory");
  }

  // ---- wave-private A staging: lane -> (row = lane>>1, half = lane&1),
  //      4 x b128 = 64B per lane per chunk (2 KB/wave, 16 KB/WG). ----
  const int ar = lane >> 1, ah = lane & 1;
  const float* asrc_row = neighbors + ((size_t)(l * N_ + n0 + ar)) * F_;
  const unsigned int adst_base =
      (unsigned int)(size_t)(void*)&Alds[(wid * 16 + ar) * AROWW + ah * 16];

  auto issueA = [&](int p) {
    const float* s = asrc_row + p * 32 + ah * 16;
    unsigned int dst = adst_base + (unsigned int)((p % NBUF) * ASLOTW) * 4u;
    #pragma unroll
    for (int j = 0; j < 4; ++j) {
      asm volatile("global_load_async_to_lds_b128 %0, %1, off"
                   :: "v"(dst + (unsigned int)(j * 16)), "v"(s + j * 4)
                   : "memory");
    }
  };

  #pragma unroll
  for (int p = 0; p < NBUF - 1; ++p) issueA(p);   // prime: chunks 0..4
  wait_async(20);            // in-order retire => all own B copies landed
  __syncthreads();           // all waves' B portions visible

  v8f acc[8];
  #pragma unroll
  for (int c = 0; c < 8; ++c)
    #pragma unroll
    for (int e = 0; e < 8; ++e) acc[c][e] = 0.f;

  auto loadB = [&](BfVec& B, int c, int t) {
    const uint4* lp =
        (const uint4*)(smem + ((size_t)((c * KCH + t) * 32 + lane)) * 8);
    *(uint4*)&B.u[0] = lp[0];
    *(uint4*)&B.u[4] = lp[1];
  };

  auto cloop = [&](const BfVec& A, int t) {        // 8 wmma, B double-buffered
    BfVec Bc, Bn;
    loadB(Bc, 0, t);
    #pragma unroll
    for (int c = 0; c < 8; ++c) {
      if (c < 7) loadB(Bn, c + 1, t);
      acc[c] = __builtin_amdgcn_wmma_f32_16x16x32_bf16(
          false, A.v, false, Bc.v, (short)0, acc[c], false, false);
      Bc = Bn;
    }
  };

  // A 16-bit 16x32 layout -> per lane: float4 @ kb, kb+4, kb+16, kb+20
  // (kb = hi8) read from this wave's rows in the LDS slot.
  auto consume = [&](int t, int waitn) {
    if (t + NBUF - 1 < NFULL) issueA(t + NBUF - 1);
    wait_async(waitn);                         // chunk t landed in LDS
    const float* as =
        Alds + (t % NBUF) * ASLOTW + (wid * 16 + lo) * AROWW + hi8;
    float4 f0 = *(const float4*)(as + 0);
    float4 f1 = *(const float4*)(as + 4);
    float4 f2 = *(const float4*)(as + 16);
    float4 f3 = *(const float4*)(as + 20);
    BfVec A;
    A.u[0] = pk_bf16(f0.x, f0.y);  A.u[1] = pk_bf16(f0.z, f0.w);
    A.u[2] = pk_bf16(f1.x, f1.y);  A.u[3] = pk_bf16(f1.z, f1.w);
    A.u[4] = pk_bf16(f2.x, f2.y);  A.u[5] = pk_bf16(f2.z, f2.w);
    A.u[6] = pk_bf16(f3.x, f3.y);  A.u[7] = pk_bf16(f3.z, f3.w);
    cloop(A, t);
  };

  for (int t = 0; t < 11; ++t) consume(t, 20);   // steady state
  consume(11, 16); consume(12, 12); consume(13, 8);   // drain
  consume(14, 4);  consume(15, 0);

  { // masked K tail (k = 512..531) via direct global loads
    const float* arow = neighbors + ((size_t)(l * N_ + n0 + lo)) * F_;
    const int kb = 512 + hi8;
    float4 f0 = {0.f,0.f,0.f,0.f}, f1 = f0, f2 = f0, f3 = f0;
    if (kb + 3  < F_) f0 = *(const float4*)(arow + kb);
    if (kb + 7  < F_) f1 = *(const float4*)(arow + kb + 4);
    if (kb + 19 < F_) f2 = *(const float4*)(arow + kb + 16);
    if (kb + 23 < F_) f3 = *(const float4*)(arow + kb + 20);
    BfVec A;
    A.u[0] = pk_bf16(f0.x, f0.y);  A.u[1] = pk_bf16(f0.z, f0.w);
    A.u[2] = pk_bf16(f1.x, f1.y);  A.u[3] = pk_bf16(f1.z, f1.w);
    A.u[4] = pk_bf16(f2.x, f2.y);  A.u[5] = pk_bf16(f2.z, f2.w);
    A.u[6] = pk_bf16(f3.x, f3.y);  A.u[7] = pk_bf16(f3.z, f3.w);
    cloop(A, 16);
  }

  // Epilogue.  C/D layout: VGPR r, lanes0-15 -> (M=r, N=lane);
  // lanes16-31 -> (M=8+r, N=lane-16).
  const int aid = (l < L_ - 1) ? action_ids[l] : -1;
  #pragma unroll
  for (int c = 0; c < 8; ++c) {
    const int d = c * 16 + lo;
    const float bv = b_fa[d];
    float m = -INFINITY;
    #pragma unroll
    for (int r = 0; r < 8; ++r) {
      float val = acc[c][r] + bv;
      m = fmaxf(m, val);
      int grow = hi8 + r;
      if (l == L_ - 1) hnts7[(size_t)(n0 + grow) * D_ + d] = val;
      if (aid == n0 + grow) hsel[l * D_ + d] = val;
    }
    m = fmaxf(m, __shfl_xor(m, 16, 32));    // merge the two lane halves
    if (hi == 0) wavemax[wid * 128 + d] = m;
  }
  __syncthreads();
  if (tid < 128) {
    float m = wavemax[tid];
    #pragma unroll
    for (int w = 1; w < 8; ++w) m = fmaxf(m, wavemax[w * 128 + tid]);
    wgmax[((size_t)l * 256 + wgi) * 128 + tid] = m;
  }
}

// ---------------------------------------------------------------------------
// k2: reduce per-WG maxima -> hmax (L, D)
// ---------------------------------------------------------------------------
__global__ void k2_hmax(const float* __restrict__ wgmax,
                        float* __restrict__ hmax) {
  int i = blockIdx.x * blockDim.x + threadIdx.x;   // l*128 + d
  if (i >= L_ * D_) return;
  int l = i >> 7, d = i & 127;
  const float* p = wgmax + ((size_t)l * 256) * 128 + d;
  float m = -INFINITY;
  for (int w = 0; w < 256; ++w) m = fmaxf(m, p[w * 128]);
  hmax[i] = m;
}

// ---------------------------------------------------------------------------
// k3: sequential GRU (8 steps) + hSt + u0, single workgroup.
// ---------------------------------------------------------------------------
__global__ void __launch_bounds__(512)
k3_gru(const float* __restrict__ query, const float* __restrict__ path_nodes,
       const float* __restrict__ W_ih, const float* __restrict__ W_hh,
       const float* __restrict__ b_ih, const float* __restrict__ b_hh,
       const float* __restrict__ W_fs, const float* __restrict__ b_fs,
       const float* __restrict__ W_fp, const float* __restrict__ b_fp,
       const float* __restrict__ hmax, const float* __restrict__ hsel,
       float* __restrict__ hSt_out, float* __restrict__ u0_out)
{
  __shared__ float h[H_];
  __shared__ float hn[H_];
  __shared__ float x[3 * D_];
  __shared__ float hst[D_];
  const int tid = threadIdx.x;

  for (int j = tid; j < H_; j += 512) h[j] = query[j];

  for (int s = 0; s < L_; ++s) {
    if (s == 0) {
      for (int j = tid; j < 2 * D_; j += 512) x[j] = 0.f;
      for (int j = tid; j < D_; j += 512) x[2 * D_ + j] = path_nodes[j];
    } else {
      for (int j = tid; j < D_; j += 512) {
        x[j]          = hmax[(s - 1) * D_ + j];
        x[D_ + j]     = hsel[(s - 1) * D_ + j];
        x[2 * D_ + j] = path_nodes[s * D_ + j];
      }
    }
    __syncthreads();
    for (int j = tid; j < H_; j += 512) {
      float gir = b_ih[j], giz = b_ih[H_ + j], gin = b_ih[2 * H_ + j];
      const float* wr = W_ih + (size_t)j * (3 * D_);
      const float* wz = W_ih + (size_t)(H_ + j) * (3 * D_);
      const float* wn = W_ih + (size_t)(2 * H_ + j) * (3 * D_);
      for (int k = 0; k < 3 * D_; ++k) {
        float xv = x[k];
        gir += wr[k] * xv; giz += wz[k] * xv; gin += wn[k] * xv;
      }
      float ghr = b_hh[j], ghz = b_hh[H_ + j], ghn = b_hh[2 * H_ + j];
      const float* vr = W_hh + (size_t)j * H_;
      const float* vz = W_hh + (size_t)(H_ + j) * H_;
      const float* vn = W_hh + (size_t)(2 * H_ + j) * H_;
      for (int k = 0; k < H_; ++k) {
        float hv = h[k];
        ghr += vr[k] * hv; ghz += vz[k] * hv; ghn += vn[k] * hv;
      }
      float r = 1.f / (1.f + __expf(-(gir + ghr)));
      float z = 1.f / (1.f + __expf(-(giz + ghz)));
      float n = tanhf(gin + r * ghn);
      hn[j] = (1.f - z) * n + z * h[j];
    }
    __syncthreads();
    for (int j = tid; j < H_; j += 512) h[j] = hn[j];
    __syncthreads();
  }

  if (tid < D_) {
    float a = b_fs[tid];
    const float* w = W_fs + (size_t)tid * H_;
    for (int k = 0; k < H_; ++k) a += w[k] * h[k];
    hst[tid] = a;
    hSt_out[tid] = a;
  }
  __syncthreads();
  if (tid == 0) {
    float u = b_fp[0];
    for (int k = 0; k < D_; ++k) u += W_fp[k] * hst[k];
    for (int k = 0; k < D_; ++k) u += W_fp[D_ + k] * hmax[(L_ - 1) * D_ + k];
    u0_out[0] = u;
  }
}

// ---------------------------------------------------------------------------
// k4: uks[n] = hSt . hnts7[n, :]
// ---------------------------------------------------------------------------
__global__ void __launch_bounds__(256)
k4_uks(const float* __restrict__ hnts7, const float* __restrict__ hSt,
       float* __restrict__ uks) {
  __shared__ float hs[D_];
  if (threadIdx.x < D_) hs[threadIdx.x] = hSt[threadIdx.x];
  __syncthreads();
  int n = blockIdx.x * blockDim.x + threadIdx.x;
  const float4* row = (const float4*)(hnts7 + (size_t)n * D_);
  float a = 0.f;
  #pragma unroll
  for (int k = 0; k < 32; ++k) {
    float4 f = row[k];
    a += f.x * hs[4*k] + f.y * hs[4*k+1] + f.z * hs[4*k+2] + f.w * hs[4*k+3];
  }
  uks[n] = a;
}

// ---------------------------------------------------------------------------
// k5: softmax over [u0, uks] -> out (32769), single workgroup.
// ---------------------------------------------------------------------------
__global__ void __launch_bounds__(1024)
k5_softmax(const float* __restrict__ u0, const float* __restrict__ uks,
           float* __restrict__ out) {
  __shared__ float red[1024];
  const int tid = threadIdx.x;
  const int M = N_ + 1;
  float m = -INFINITY;
  for (int i = tid; i < M; i += 1024) {
    float v = (i == 0) ? u0[0] : uks[i - 1];
    m = fmaxf(m, v);
  }
  red[tid] = m; __syncthreads();
  for (int s = 512; s > 0; s >>= 1) {
    if (tid < s) red[tid] = fmaxf(red[tid], red[tid + s]);
    __syncthreads();
  }
  float gmax = red[0];
  __syncthreads();
  float sum = 0.f;
  for (int i = tid; i < M; i += 1024) {
    float v = (i == 0) ? u0[0] : uks[i - 1];
    float e = __expf(v - gmax);
    out[i] = e;
    sum += e;
  }
  red[tid] = sum; __syncthreads();
  for (int s = 512; s > 0; s >>= 1) {
    if (tid < s) red[tid] += red[tid + s];
    __syncthreads();
  }
  float inv = 1.f / red[0];
  __syncthreads();
  for (int i = tid; i < M; i += 1024) out[i] *= inv;
}

// ---------------------------------------------------------------------------
extern "C" void kernel_launch(void* const* d_in, const int* in_sizes, int n_in,
                              void* d_out, int out_size, void* d_ws,
                              size_t ws_size, hipStream_t stream) {
  const float* query      = (const float*)d_in[0];
  const float* path_nodes = (const float*)d_in[1];
  const float* neighbors  = (const float*)d_in[2];
  const int*   action_ids = (const int*)  d_in[3];
  const float* W_ih = (const float*)d_in[4];
  const float* W_hh = (const float*)d_in[5];
  const float* b_ih = (const float*)d_in[6];
  const float* b_hh = (const float*)d_in[7];
  const float* W_fa = (const float*)d_in[8];
  const float* b_fa = (const float*)d_in[9];
  const float* W_fs = (const float*)d_in[10];
  const float* b_fs = (const float*)d_in[11];
  const float* W_fp = (const float*)d_in[12];
  const float* b_fp = (const float*)d_in[13];
  float* out = (float*)d_out;

  // Workspace layout (floats / words), total ~18.2 MB.
  float* ws = (float*)d_ws;
  unsigned int* Bp = (unsigned int*)ws;      // BWORDS words (packed bf16 W_fa)
  float* wgmax = ws + BWORDS;                // L*256*128 = 262144
  float* hmax  = wgmax + 262144;             // 1024
  float* hsel  = hmax + 1024;                // 896
  float* hnts7 = hsel + 896;                 // N*D = 4194304 (16B-aligned)
  float* hSt   = hnts7 + 4194304;            // 128
  float* u0    = hSt + 128;                  // 1
  float* uks   = u0 + 1;                     // 32768

  // Allow >64KB dynamic LDS for k1 (LDS is 320KB/WGP on CDNA5).
  hipFuncSetAttribute((const void*)k1_gemm,
                      hipFuncAttributeMaxDynamicSharedMemorySize,
                      K1_SMEM_BYTES);

  k0_pack_wfa<<<(BWORDS + 255) / 256, 256, 0, stream>>>(W_fa, Bp);
  k1_gemm<<<L_ * WGS_PER_HOP, 256, K1_SMEM_BYTES, stream>>>(
      neighbors, b_fa, action_ids, Bp, wgmax, hsel, hnts7);
  k2_hmax<<<(L_ * D_ + 255) / 256, 256, 0, stream>>>(wgmax, hmax);
  k3_gru<<<1, 512, 0, stream>>>(query, path_nodes, W_ih, W_hh, b_ih, b_hh,
                                W_fs, b_fs, W_fp, b_fp, hmax, hsel, hSt, u0);
  k4_uks<<<N_ / 256, 256, 0, stream>>>(hnts7, hSt, uks);
  k5_softmax<<<1, 1024, 0, stream>>>(u0, uks, out);
}
